// VisionMambaBlock_25795573580036
// MI455X (gfx1250) — compile-verified, matrix operand order checked by reading, not run on hardware
//
#include <hip/hip_runtime.h>
#include <cstdint>

#define DEV __device__ __forceinline__

typedef unsigned short u16;
typedef unsigned int   u32;
typedef unsigned long long u64;

typedef __attribute__((ext_vector_type(16))) __bf16 v16bf;
typedef __attribute__((ext_vector_type(8)))  float  v8f;
typedef __attribute__((ext_vector_type(4)))  u32    u32x4;
typedef __attribute__((ext_vector_type(4)))  int    i32x4;

constexpr int Bb   = 8;
constexpr int Hh   = 56;
constexpr int Ww   = 56;
constexpr int Cc   = 192;
constexpr int Ln   = Hh * Ww;        // 3136
constexpr int Din  = 384;
constexpr int Rr   = 12;
constexpr int Hid  = 768;
constexpr int Mrow = Bb * Ln;        // 25088
constexpr int Ecnt = 2 * 56 * 55;    // 6160 grid edges
constexpr int EPAD = 8192;           // bitonic pad
constexpr int CH   = 16;             // channels per tree-scan block (LDS slice)

#if defined(__HIP_DEVICE_COMPILE__) && \
    __has_builtin(__builtin_amdgcn_global_load_async_to_lds_b128) && \
    __has_builtin(__builtin_amdgcn_s_wait_asynccnt)
#define USE_ASYNC_LDS 1
typedef __attribute__((address_space(1))) i32x4* gbl_v4;   // global (AS1) v4i32*
typedef __attribute__((address_space(3))) i32x4* lds_v4;   // LDS (AS3) v4i32*
#else
#define USE_ASYNC_LDS 0
#endif

DEV u16 f2bf(float f) {
  u32 u = __float_as_uint(f);
  u32 r = u + 0x7FFFu + ((u >> 16) & 1u);   // round-to-nearest-even
  return (u16)(r >> 16);
}

// ---------------------------------------------------------------------------
// Per-image MST (Kruskal, cosine-distance keys) + BFS.  One block per image.
// ---------------------------------------------------------------------------
__global__ __launch_bounds__(512) void mst_bfs_kernel(
    const float* __restrict__ X,   // [B, L, C]
    u64* __restrict__ keys,        // [B, EPAD] scratch
    int* __restrict__ adj,         // [B, L*4] scratch
    int* __restrict__ bfsIdx,      // [B, L]
    int* __restrict__ bfsPar)      // [B, L]
{
  __shared__ float sNorm[Ln];
  __shared__ int   sUF[Ln];
  __shared__ int   sDeg[Ln];
  const int b   = blockIdx.x;
  const int tid = threadIdx.x;
  const float* f = X + (size_t)b * Ln * Cc;
  u64* K = keys + (size_t)b * EPAD;
  int* A = adj  + (size_t)b * Ln * 4;

  // 1) node norms
  for (int n = tid; n < Ln; n += 512) {
    const float* p = f + (size_t)n * Cc;
    float s = 0.f;
    for (int c = 0; c < Cc; ++c) s += p[c] * p[c];
    sNorm[n] = sqrtf(s) + 1e-8f;
  }
  __syncthreads();

  // 2) edge keys: weight = 1-cos (exp() is monotone -> same order as reference)
  for (int e = tid; e < EPAD; e += 512) {
    u64 kv = 0xFFFFFFFFFFFFFFFFull;
    if (e < Ecnt) {
      int u, v;
      if (e < 3080) { int r = e / 55, c = e % 55; u = r * 56 + c; v = u + 1; }
      else          { u = e - 3080; v = u + 56; }
      const float* pu = f + (size_t)u * Cc;
      const float* pv = f + (size_t)v * Cc;
      float d = 0.f;
      for (int c = 0; c < Cc; ++c) d += pu[c] * pv[c];
      float key = 1.0f - d / (sNorm[u] * sNorm[v]);
      u32 kb = __float_as_uint(key);
      kb ^= (kb >> 31) ? 0xFFFFFFFFu : 0x80000000u;  // float -> sortable uint
      kv = ((u64)kb << 32) | (u32)e;                 // idx tiebreak = stable
    }
    K[e] = kv;
  }
  __syncthreads();

  // 3) bitonic sort ascending over EPAD u64
  for (int k = 2; k <= EPAD; k <<= 1) {
    for (int j = k >> 1; j > 0; j >>= 1) {
      for (int t = tid; t < EPAD; t += 512) {
        int ixj = t ^ j;
        if (ixj > t) {
          u64 a = K[t], c = K[ixj];
          bool asc = ((t & k) == 0);
          if ((a > c) == asc) { K[t] = c; K[ixj] = a; }
        }
      }
      __syncthreads();
    }
  }

  // 4) init union-find / degree / visited
  int* vis = (int*)sNorm;   // norms no longer needed
  for (int n = tid; n < Ln; n += 512) { sUF[n] = n; sDeg[n] = 0; vis[n] = 0; }
  __syncthreads();

  // 5) serial Kruskal + BFS on one lane
  if (tid == 0) {
    int cnt = 0;
    for (int e = 0; e < Ecnt && cnt < Ln - 1; ++e) {
      int idx = (int)(u32)(K[e] & 0xFFFFFFFFu);
      int u, v;
      if (idx < 3080) { int r = idx / 55, c = idx % 55; u = r * 56 + c; v = u + 1; }
      else            { u = idx - 3080; v = u + 56; }
      int ru = u; while (sUF[ru] != ru) { sUF[ru] = sUF[sUF[ru]]; ru = sUF[ru]; }
      int rv = v; while (sUF[rv] != rv) { sUF[rv] = sUF[sUF[rv]]; rv = sUF[rv]; }
      if (ru != rv) {
        sUF[ru] = rv;
        A[u * 4 + sDeg[u]++] = v;
        A[v * 4 + sDeg[v]++] = u;
        ++cnt;
      }
    }
    int* bi = bfsIdx + (size_t)b * Ln;
    int* bp = bfsPar + (size_t)b * Ln;
    bi[0] = 0; bp[0] = 0; vis[0] = 1;
    int head = 0, tail = 1;
    while (head < tail) {
      int u = bi[head];
      int dg = sDeg[u];
      for (int d = 0; d < dg; ++d) {
        int v = A[u * 4 + d];
        if (!vis[v]) { vis[v] = 1; bi[tail] = v; bp[tail] = u; ++tail; }
      }
      ++head;
    }
  }
}

// ---------------------------------------------------------------------------
// LayerNorm over 192 channels, one wave per row (wave32).
// ---------------------------------------------------------------------------
__global__ __launch_bounds__(256) void ln192_kernel(
    const float* __restrict__ X, const float* __restrict__ w,
    const float* __restrict__ bsh, u16* __restrict__ outBf,
    float* __restrict__ outF, int rows)
{
  int wave = threadIdx.x >> 5, lane = threadIdx.x & 31;
  int row = blockIdx.x * 8 + wave;
  if (row >= rows) return;
  const float* x = X + (size_t)row * Cc;
  float xv[6];
  float s = 0.f, s2 = 0.f;
#pragma unroll
  for (int t = 0; t < 6; ++t) {
    float v = x[lane + 32 * t];
    xv[t] = v; s += v; s2 += v * v;
  }
#pragma unroll
  for (int m = 16; m >= 1; m >>= 1) { s += __shfl_xor(s, m, 32); s2 += __shfl_xor(s2, m, 32); }
  float mu  = s * (1.f / Cc);
  float var = s2 * (1.f / Cc) - mu * mu;
  float rs  = rsqrtf(var + 1e-5f);
#pragma unroll
  for (int t = 0; t < 6; ++t) {
    int j = lane + 32 * t;
    float o = (xv[t] - mu) * rs * w[j] + bsh[j];
    if (outBf) outBf[(size_t)row * Cc + j] = f2bf(o);
    else       outF [(size_t)row * Cc + j] = o;
  }
}

// ---------------------------------------------------------------------------
// bf16 WMMA GEMM:  C[M,N] = A[M,K] * W[N,K]^T  (both operands K-contiguous)
// block tile 128x64, 8 waves (4x2), wave tile 32x32 (2x2 fragments).
// EPI: 0 = f32 store, 1 = +resid f32, 2 = +bias,gelu -> bf16, 3 = +bias+resid f32
// ---------------------------------------------------------------------------
DEV v16bf load_frag(const u16* __restrict__ base, int ld, int mBase, int kBase, int lane) {
  int m = mBase + (lane & 15);
  int h = lane >> 4;
  const u16* p = base + (size_t)m * ld + kBase + 8 * h;
  union { u32x4 q[2]; v16bf v; } fr;
  fr.q[0] = *(const u32x4*)(p);
  fr.q[1] = *(const u32x4*)(p + 16);
  return fr.v;
}

template <int EPI>
__global__ __launch_bounds__(256) void gemm_bf16_kernel(
    const u16* __restrict__ Am, const u16* __restrict__ Bm,
    float* __restrict__ Cf, u16* __restrict__ Cbf,
    const float* __restrict__ bias, const float* __restrict__ resid,
    int N, int K)
{
  const int lane = threadIdx.x & 31;
  const int wave = threadIdx.x >> 5;
  const int wm = wave >> 1, wn = wave & 1;
  const int mW = blockIdx.x * 128 + wm * 32;
  const int nW = blockIdx.y * 64 + wn * 32;

  v8f acc[2][2];
#pragma unroll
  for (int i = 0; i < 2; ++i)
#pragma unroll
    for (int j = 0; j < 2; ++j) {
      v8f z = {0.f, 0.f, 0.f, 0.f, 0.f, 0.f, 0.f, 0.f};
      acc[i][j] = z;
    }

  for (int k = 0; k < K; k += 32) {
    v16bf a0 = load_frag(Am, K, mW,      k, lane);
    v16bf a1 = load_frag(Am, K, mW + 16, k, lane);
    v16bf b0 = load_frag(Bm, K, nW,      k, lane);
    v16bf b1 = load_frag(Bm, K, nW + 16, k, lane);
    if (k + 32 < K) {
      __builtin_prefetch(Am + (size_t)(mW + (lane & 15)) * K + k + 32, 0, 1);
      __builtin_prefetch(Bm + (size_t)(nW + (lane & 15)) * K + k + 32, 0, 1);
    }
    acc[0][0] = __builtin_amdgcn_wmma_f32_16x16x32_bf16(false, a0, false, b0, (short)0, acc[0][0], false, false);
    acc[0][1] = __builtin_amdgcn_wmma_f32_16x16x32_bf16(false, a0, false, b1, (short)0, acc[0][1], false, false);
    acc[1][0] = __builtin_amdgcn_wmma_f32_16x16x32_bf16(false, a1, false, b0, (short)0, acc[1][0], false, false);
    acc[1][1] = __builtin_amdgcn_wmma_f32_16x16x32_bf16(false, a1, false, b1, (short)0, acc[1][1], false, false);
  }

  const int half  = lane >> 4;
  const int nLane = lane & 15;
#pragma unroll
  for (int mi = 0; mi < 2; ++mi) {
#pragma unroll
    for (int ni = 0; ni < 2; ++ni) {
      int n = nW + ni * 16 + nLane;
#pragma unroll
      for (int r = 0; r < 8; ++r) {
        int m = mW + mi * 16 + r + 8 * half;   // C/D layout: VGPR r, lane half
        size_t off = (size_t)m * N + n;
        float v = acc[mi][ni][r];
        if (EPI == 0) {
          Cf[off] = v;
        } else if (EPI == 1) {
          Cf[off] = v + resid[off];
        } else if (EPI == 2) {
          float t = v + bias[n];
          t = 0.5f * t * (1.f + erff(t * 0.70710678118f));  // exact gelu
          Cbf[off] = f2bf(t);
        } else {
          Cf[off] = v + bias[n] + resid[off];
        }
      }
    }
  }
}

// ---------------------------------------------------------------------------
// Depthwise 3x3 conv (SAME) on xs = xz[..., :384], + bias, silu -> u (f32)
// ---------------------------------------------------------------------------
__global__ __launch_bounds__(256) void conv_silu_kernel(
    const float* __restrict__ xz, const float* __restrict__ cw,
    const float* __restrict__ cb, float* __restrict__ u)
{
  int idx = blockIdx.x * 256 + threadIdx.x;
  if (idx >= Bb * Ln * Din) return;
  int c = idx % Din;
  int l = (idx / Din) % Ln;
  int b = idx / (Din * Ln);
  int h = l / Ww, w = l % Ww;
  float s = cb[c];
#pragma unroll
  for (int kh = 0; kh < 3; ++kh) {
    int hh = h + kh - 1;
    if ((unsigned)hh >= (unsigned)Hh) continue;
#pragma unroll
    for (int kw = 0; kw < 3; ++kw) {
      int w2 = w + kw - 1;
      if ((unsigned)w2 >= (unsigned)Ww) continue;
      s += xz[((size_t)(b * Ln + hh * Ww + w2)) * (2 * Din) + c] * cw[(kh * 3 + kw) * Din + c];
    }
  }
  u[idx] = s / (1.f + expf(-s));
}

// x_dbl = u @ x_proj_w^T   (N=14, tiny -> VALU)
__global__ __launch_bounds__(256) void xproj_kernel(
    const float* __restrict__ u, const float* __restrict__ wp, float* __restrict__ xdbl)
{
  int idx = blockIdx.x * 256 + threadIdx.x;
  if (idx >= Mrow * 14) return;
  int j = idx % 14, m = idx / 14;
  const float* ur = u  + (size_t)m * Din;
  const float* wr = wp + (size_t)j * Din;
  float s = 0.f;
  for (int k = 0; k < Din; ++k) s += ur[k] * wr[k];
  xdbl[idx] = s;
}

// dts = softplus(x_dbl[:, :12] @ dt_w^T + dt_b);  w = exp(dts*A);  S = dts*Bs*u
__global__ __launch_bounds__(256) void dt_w_feat_kernel(
    const float* __restrict__ xdbl, const float* __restrict__ dtw,
    const float* __restrict__ dtb, const float* __restrict__ alog,
    const float* __restrict__ u, float* __restrict__ Wb, float* __restrict__ Sb)
{
  int idx = blockIdx.x * 256 + threadIdx.x;
  if (idx >= Mrow * Din) return;
  int d = idx % Din, m = idx / Din;
  const float* xr = xdbl + (size_t)m * 14;
  float dtr = dtb[d];
#pragma unroll
  for (int r = 0; r < Rr; ++r) dtr += xr[r] * dtw[d * Rr + r];
  float dts = (dtr > 20.f) ? dtr : log1pf(expf(dtr));
  float Aa  = -expf(alog[d]);
  Wb[idx] = expf(dts * Aa);
  Sb[idx] = dts * xr[Rr] * u[idx];     // xr[12] = Bs
}

// ---------------------------------------------------------------------------
// Tree scan with the serial dependency chain routed through LDS.
// Each block owns a [Ln][CH] f32 slice of S (196 KB dynamic LDS) for one
// (image, channel-group).  The leaves->root and root->leaves passes then
// bounce store->load through LDS (ds ops) instead of L2, cutting the
// dependent-iteration latency by ~an order of magnitude.  w[n] reads stay
// in global: their addresses don't depend on the chain, so they pipeline.
// S slice staged in via gfx1250 async-to-LDS when the builtins exist.
// ---------------------------------------------------------------------------
__global__ __launch_bounds__(128) void tree_scan_kernel(
    const int* __restrict__ bfsIdx, const int* __restrict__ bfsPar,
    const float* __restrict__ Wb, float* __restrict__ Sb)
{
  __shared__ int sI[Ln];
  __shared__ int sP[Ln];
  extern __shared__ float sS[];                 // [Ln][CH]
  const int groups = Din / CH;                  // 24
  const int b     = blockIdx.x / groups;
  const int cBase = (blockIdx.x % groups) * CH;
  const int tid   = threadIdx.x;
  const size_t base = (size_t)b * Ln * Din;

  for (int i = tid; i < Ln; i += 128) { sI[i] = bfsIdx[b * Ln + i]; sP[i] = bfsPar[b * Ln + i]; }

  // stage S slice: global [n][Din] strided -> LDS [n][CH] packed (16B chunks)
#if USE_ASYNC_LDS
  {
    const float* gs = Sb + base + cBase;
    for (int q = tid; q < Ln * (CH / 4); q += 128) {
      int n = q >> 2, part = q & 3;
      const float* gp = gs + (size_t)n * Din + part * 4;
      __builtin_amdgcn_global_load_async_to_lds_b128(
          (gbl_v4)(uintptr_t)gp,
          (lds_v4)(u32)(uintptr_t)&sS[q * 4], 0, 0);
    }
    __builtin_amdgcn_s_wait_asynccnt(0);
  }
#else
  for (int q = tid; q < Ln * CH; q += 128) {
    int n = q / CH, c = q % CH;
    sS[q] = Sb[base + (size_t)n * Din + cBase + c];
  }
#endif
  __syncthreads();

  if (tid < CH) {
    const int c = tid;
    const float* Wg = Wb + base + cBase + c;
    for (int i = Ln - 1; i >= 1; --i) {         // leaves -> root
      int n = sI[i], p = sP[i];
      sS[p * CH + c] += Wg[(size_t)n * Din] * sS[n * CH + c];
    }
    for (int i = 1; i < Ln; ++i) {              // root -> leaves, in place
      int n = sI[i], p = sP[i];
      float wn = Wg[(size_t)n * Din];
      float sn = sS[n * CH + c];
      sS[n * CH + c] = sn + wn * (sS[p * CH + c] - wn * sn);
    }
  }
  __syncthreads();

  // write back
  for (int q = tid; q < Ln * CH; q += 128) {
    int n = q / CH, c = q % CH;
    Sb[base + (size_t)n * Din + cBase + c] = sS[q];
  }
}

// y = Cs*H + Ds*u; LayerNorm(384); * silu(z); -> bf16.  One wave per row.
__global__ __launch_bounds__(256) void gate_kernel(
    const float* __restrict__ Hbuf, const float* __restrict__ u,
    const float* __restrict__ xdbl, const float* __restrict__ Ds,
    const float* __restrict__ onw, const float* __restrict__ onb,
    const float* __restrict__ xz, u16* __restrict__ ybf)
{
  int wave = threadIdx.x >> 5, lane = threadIdx.x & 31;
  int row = blockIdx.x * 8 + wave;
  if (row >= Mrow) return;
  float Csv = xdbl[(size_t)row * 14 + 13];
  float yv[12];
  float s = 0.f, s2 = 0.f;
#pragma unroll
  for (int t = 0; t < 12; ++t) {
    int j = lane + 32 * t;
    float y = Csv * Hbuf[(size_t)row * Din + j] + Ds[j] * u[(size_t)row * Din + j];
    yv[t] = y; s += y; s2 += y * y;
  }
#pragma unroll
  for (int m = 16; m >= 1; m >>= 1) { s += __shfl_xor(s, m, 32); s2 += __shfl_xor(s2, m, 32); }
  float mu  = s * (1.f / Din);
  float var = s2 / Din - mu * mu;
  float rs  = rsqrtf(var + 1e-5f);
#pragma unroll
  for (int t = 0; t < 12; ++t) {
    int j = lane + 32 * t;
    float o = (yv[t] - mu) * rs * onw[j] + onb[j];
    float z = xz[(size_t)row * (2 * Din) + Din + j];
    o *= z / (1.f + expf(-z));
    ybf[(size_t)row * Din + j] = f2bf(o);
  }
}

__global__ __launch_bounds__(256) void f32_to_bf16_kernel(
    const float* __restrict__ src, u16* __restrict__ dst, int n)
{
  int i = blockIdx.x * 256 + threadIdx.x;
  if (i < n) dst[i] = f2bf(src[i]);
}

// ---------------------------------------------------------------------------
extern "C" void kernel_launch(void* const* d_in, const int* in_sizes, int n_in,
                              void* d_out, int out_size, void* d_ws, size_t ws_size,
                              hipStream_t stream)
{
  (void)in_sizes; (void)n_in; (void)out_size; (void)ws_size;
  const float* x0   = (const float*)d_in[0];
  const float* n1w  = (const float*)d_in[1];
  const float* n1b  = (const float*)d_in[2];
  const float* ipw  = (const float*)d_in[3];
  const float* cw   = (const float*)d_in[4];
  const float* cb   = (const float*)d_in[5];
  const float* xpw  = (const float*)d_in[6];
  const float* dtw  = (const float*)d_in[7];
  const float* dtb  = (const float*)d_in[8];
  const float* alog = (const float*)d_in[9];
  const float* DsP  = (const float*)d_in[10];
  const float* onw  = (const float*)d_in[11];
  const float* onb  = (const float*)d_in[12];
  const float* opw  = (const float*)d_in[13];
  const float* n2w  = (const float*)d_in[14];
  const float* n2b  = (const float*)d_in[15];
  const float* f1w  = (const float*)d_in[16];
  const float* f1b  = (const float*)d_in[17];
  const float* f2w  = (const float*)d_in[18];
  const float* f2b  = (const float*)d_in[19];
  const float* fnw  = (const float*)d_in[20];
  const float* fnb  = (const float*)d_in[21];

  char* ws = (char*)d_ws;
  size_t cur = 0;
  auto alloc = [&](size_t bytes) -> char* {
    char* p = ws + cur;
    cur = (cur + bytes + 255) & ~(size_t)255;
    return p;
  };

  float* xz   = (float*)alloc((size_t)Mrow * 768 * 4);   // also reused as fc1 bf16 out
  float* uF   = (float*)alloc((size_t)Mrow * Din * 4);
  float* xdbl = (float*)alloc((size_t)Mrow * 14 * 4);
  float* Wb   = (float*)alloc((size_t)Mrow * Din * 4);
  float* Sb   = (float*)alloc((size_t)Mrow * Din * 4);
  u16*   ybf  = (u16*)  alloc((size_t)Mrow * Din * 2);
  float* x2   = (float*)alloc((size_t)Mrow * Cc * 4);
  u16*   hbf  = (u16*)  alloc((size_t)Mrow * Cc * 2);
  u16*   xnbf = (u16*)  alloc((size_t)Mrow * Cc * 2);
  float* xA   = (float*)alloc((size_t)Mrow * Cc * 4);
  u16*   ipwb = (u16*)  alloc((size_t)Hid * Cc * 2);
  u16*   opwb = (u16*)  alloc((size_t)Cc * Din * 2);
  u16*   f1wb = (u16*)  alloc((size_t)Hid * Cc * 2);
  u16*   f2wb = (u16*)  alloc((size_t)Cc * Hid * 2);
  u64*   keys = (u64*)  alloc((size_t)Bb * EPAD * 8);
  int*   adjB = (int*)  alloc((size_t)Bb * Ln * 4 * 4);
  int*   bfsI = (int*)  alloc((size_t)Bb * Ln * 4);
  int*   bfsP = (int*)  alloc((size_t)Bb * Ln * 4);
  u16*   fc1bf = (u16*)xz;   // xz is dead after gate_kernel consumes z

  constexpr size_t SCAN_LDS = (size_t)Ln * CH * 4;   // 196 KB dynamic LDS

  for (int layer = 0; layer < 2; ++layer) {
    const float* xin = (layer == 0) ? x0 : xA;
    float* xout = (layer == 0) ? xA : (float*)d_out;

    // weight f32 -> bf16
    f32_to_bf16_kernel<<<(Hid * Cc + 255) / 256, 256, 0, stream>>>(ipw + (size_t)layer * Hid * Cc, ipwb, Hid * Cc);
    f32_to_bf16_kernel<<<(Cc * Din + 255) / 256, 256, 0, stream>>>(opw + (size_t)layer * Cc * Din, opwb, Cc * Din);
    f32_to_bf16_kernel<<<(Hid * Cc + 255) / 256, 256, 0, stream>>>(f1w + (size_t)layer * Hid * Cc, f1wb, Hid * Cc);
    f32_to_bf16_kernel<<<(Cc * Hid + 255) / 256, 256, 0, stream>>>(f2w + (size_t)layer * Cc * Hid, f2wb, Cc * Hid);

    // tree from current x
    mst_bfs_kernel<<<Bb, 512, 0, stream>>>(xin, keys, adjB, bfsI, bfsP);

    // norm1 -> bf16; in_proj GEMM (M x 768, K=192)
    ln192_kernel<<<Mrow / 8, 256, 0, stream>>>(xin, n1w + layer * Cc, n1b + layer * Cc, xnbf, nullptr, Mrow);
    gemm_bf16_kernel<0><<<dim3(Mrow / 128, Hid / 64), 256, 0, stream>>>(
        xnbf, ipwb, xz, nullptr, nullptr, nullptr, Hid, Cc);

    // depthwise conv + silu -> u
    conv_silu_kernel<<<(Bb * Ln * Din + 255) / 256, 256, 0, stream>>>(
        xz, cw + layer * 9 * Din, cb + layer * Din, uF);

    // x_proj, dt_proj + softplus, w/feat
    xproj_kernel<<<(Mrow * 14 + 255) / 256, 256, 0, stream>>>(uF, xpw + layer * 14 * Din, xdbl);
    dt_w_feat_kernel<<<(Mrow * Din + 255) / 256, 256, 0, stream>>>(
        xdbl, dtw + layer * Din * Rr, dtb + layer * Din, alog + layer * Din, uF, Wb, Sb);

    // serial tree scan, LDS-resident chain
    tree_scan_kernel<<<Bb * (Din / CH), 128, SCAN_LDS, stream>>>(bfsI, bfsP, Wb, Sb);

    // y combine + out_norm + silu gate -> bf16; out_proj GEMM + residual
    gate_kernel<<<Mrow / 8, 256, 0, stream>>>(
        Sb, uF, xdbl, DsP + layer * Din, onw + layer * Din, onb + layer * Din, xz, ybf);
    gemm_bf16_kernel<1><<<dim3(Mrow / 128, Cc / 64), 256, 0, stream>>>(
        ybf, opwb, x2, nullptr, nullptr, xin, Cc, Din);

    // MLP: norm2 -> fc1(+bias,gelu)->bf16 -> fc2(+bias,+residual)->x_next
    ln192_kernel<<<Mrow / 8, 256, 0, stream>>>(x2, n2w + layer * Cc, n2b + layer * Cc, hbf, nullptr, Mrow);
    gemm_bf16_kernel<2><<<dim3(Mrow / 128, Hid / 64), 256, 0, stream>>>(
        hbf, f1wb, nullptr, fc1bf, f1b + layer * Hid, nullptr, Hid, Cc);
    gemm_bf16_kernel<3><<<dim3(Mrow / 128, Cc / 64), 256, 0, stream>>>(
        fc1bf, f2wb, xout, nullptr, f2b + layer * Cc, x2, Cc, Hid);
  }

  // final LayerNorm, in place on d_out
  ln192_kernel<<<Mrow / 8, 256, 0, stream>>>((const float*)d_out, fnw, fnb, nullptr, (float*)d_out, Mrow);
}
